// FeatureGroupTRM_41480794145074
// MI455X (gfx1250) — matrix operand order repaired and checked by reading, not compiled, not run on hardware
//
#include <hip/hip_runtime.h>
#include <hip/hip_fp16.h>

typedef __attribute__((ext_vector_type(16))) _Float16 v16h;
typedef __attribute__((ext_vector_type(8)))  _Float16 v8h;
typedef __attribute__((ext_vector_type(8)))  float    v8f;

#define NPROPS 22
#define SDIM   6
#define M2VD   200
#define DMODEL 48
#define NHEAD  2
#define HDIM   24
#define FFD    96
#define NSTEPS 16
#define TOK    22
#define XDIM   (NPROPS*SDIM + M2VD)   /* 332 */
#define LNEPS  1e-5f

/* ---- packed f16 weight layout inside d_ws (offsets in halves) ---- */
#define OFF_TP     0                         /* 48  x 32 */
#define OFF_SAIN   (OFF_TP    + 48*32)       /* 144 x 64 */
#define OFF_SAOUT  (OFF_SAIN  + 144*64)      /* 48  x 64 */
#define OFF_FF1    (OFF_SAOUT + 48*64)       /* 96  x 64 */
#define OFF_FF2    (OFF_FF1   + 96*64)       /* 48  x 96 */
#define OFF_CE0KV  (OFF_FF2   + 48*96)       /* 96  x 64 */
#define OFF_CETKV  (OFF_CE0KV + 96*64)       /* 96  x 64 */
#define PACK_TOTAL (OFF_CETKV + 96*64)

/* d_in index map: setup_inputs dict order, params in insertion order */
enum {
  IDX_X = 0,
  IDX_TP_W, IDX_TP_B, IDX_TP_G, IDX_TP_BT,
  IDX_ZI_W, IDX_ZI_B, IDX_ZI_G, IDX_ZI_BT,
  IDX_SA_IN_W, IDX_SA_IN_B, IDX_SA_OUT_W, IDX_SA_OUT_B,
  IDX_CE0_IN_W, IDX_CE0_IN_B, IDX_CE0_OUT_W, IDX_CE0_OUT_B,
  IDX_CET_IN_W, IDX_CET_IN_B, IDX_CET_OUT_W, IDX_CET_OUT_B,
  IDX_SAN_G, IDX_SAN_BT, IDX_CE0N_G, IDX_CE0N_BT, IDX_CETN_G, IDX_CETN_BT,
  IDX_SAFF1_W, IDX_SAFF1_B, IDX_SAFF2_W, IDX_SAFF2_B, IDX_SAFFN_G, IDX_SAFFN_BT,
  IDX_ZFF1_W, IDX_ZFF1_B, IDX_ZFF2_W, IDX_ZFF2_B, IDX_ZFFN_G, IDX_ZFFN_BT,
  IDX_YFF1_W, IDX_YFF1_B, IDX_YFF2_W, IDX_YFF2_B, IDX_YFFN_G, IDX_YFFN_BT,
  IDX_HEAD_W, IDX_HEAD_B, N_IDX
};

struct KParams { const float* p[N_IDX]; };

/* wave-scope sync: all LDS traffic is intra-wave; per-wave DS ops complete
   in order on CDNA5, so we only need a compiler scheduling barrier + a
   wavefront-scope fence (no s_barrier hardware cost).                        */
__device__ inline void wsync() {
  __builtin_amdgcn_fence(__ATOMIC_RELEASE, "wavefront");
  __builtin_amdgcn_wave_barrier();
  __builtin_amdgcn_fence(__ATOMIC_ACQUIRE, "wavefront");
}

/* ---- weight packer: f32 (N,K) row slice -> f16 [N][Kpad], zero padded ---- */
__global__ void pack_w(const float* __restrict__ src, int row_off, int K, int N,
                       int Kpad, _Float16* __restrict__ dst) {
  int i = blockIdx.x * blockDim.x + threadIdx.x;
  if (i >= N * Kpad) return;
  int n = i / Kpad, k = i - n * Kpad;
  float v = (k < K) ? src[(size_t)(row_off + n) * K + k] : 0.0f;
  dst[i] = (_Float16)v;
}

/* =============================== WMMA helpers =============================== */
/* A fragment, 16x32 f16, in contiguous-K order (same as B):
   lane L: M = mbase + (L&15); 16 contiguous K values at kbase + (L>=16 ? 16 : 0)
   No guards: pad rows/cols may contain garbage; pad K-columns always multiply
   zero-padded B columns, and pad C rows land in bufC scratch rows 22..31.     */
__device__ inline v16h load_a_frag(const _Float16* A, int lda, int mbase,
                                   int kbase, int lane) {
  int m = mbase + (lane & 15);
  const _Float16* p = A + m * lda + kbase + ((lane & 16) ? 16 : 0);
  return *(const v16h*)p;
}

/* B fragment, 32x16 f16 from packed global weights W[N][Kpad]:
   lane L: N = nbase + (L&15); 16 contiguous K values at kbase + (L>=16?16:0)   */
__device__ inline v16h load_b_frag(const _Float16* W, int kpad, int nbase,
                                   int kbase, int lane) {
  const _Float16* p = W + (size_t)(nbase + (lane & 15)) * kpad + kbase +
                      ((lane & 16) ? 16 : 0);
  return *(const v16h*)p;
}

/* C/D fragment store (unconditional; C has 32-row capacity):
   lane L: N = nbase+(L&15); VGPR r -> M = mbase+(L>=16?8:0)+r                  */
__device__ inline void store_c_frag(float* C, int ldc, int mbase, int nbase,
                                    const float* bias, v8f c, int lane) {
  int n  = nbase + (lane & 15);
  int m0 = mbase + ((lane & 16) ? 8 : 0);
  float b = bias[n];
  float* p = C + m0 * ldc + n;
#pragma unroll
  for (int r = 0; r < 8; ++r) p[r * ldc] = c[r] + b;
}

/* C[32 x N] (f32 LDS, rows>=22 scratch) = A[22 x K] (f16 LDS) * W^T + bias.
   A fragments (2 M-tiles x <=3 K-tiles) hoisted into registers once; each B
   fragment loaded once and feeds both M-tiles.                                */
__device__ inline void mm_wmma(const _Float16* A, int lda, int K,
                               const _Float16* W, int kpad, int N,
                               const float* bias, float* C, int ldc, int lane) {
  const int kt = (K + 31) >> 5;          /* constant after inlining */
  v16h a[2][3];
#pragma unroll
  for (int mb = 0; mb < 2; ++mb)
    for (int k = 0; k < kt; ++k)
      a[mb][k] = load_a_frag(A, lda, mb * 16, k * 32, lane);

  for (int nb = 0; nb < N; nb += 16) {
    v16h b[3];
    for (int k = 0; k < kt; ++k) b[k] = load_b_frag(W, kpad, nb, k * 32, lane);
#pragma unroll
    for (int mb = 0; mb < 2; ++mb) {
      v8f c = {};
      for (int k = 0; k < kt; ++k)
        c = __builtin_amdgcn_wmma_f32_16x16x32_f16(false, a[mb][k], false, b[k],
                                                   (short)0, c, false, false);
      store_c_frag(C, ldc, mb * 16, nb, bias, c, lane);
    }
  }
}

/* =============================== VALU helpers =============================== */
__device__ inline float gelu_f(float x) {
  return 0.5f * x * (1.0f + erff(x * 0.70710678118654752f));
}

/* per-row LayerNorm(+optional GELU) over `cols`, lane = row, write f16 dst    */
__device__ inline void rows_ln_f16(const float* S, int lds_, int cols,
                                   const float* g, const float* bt, bool dogelu,
                                   _Float16* D, int ldd, int lane) {
  if (lane < TOK) {
    const float* r = S + lane * lds_;
    float m = 0.f, q = 0.f;
    for (int c = 0; c < cols; ++c) { float v = r[c]; m += v; q += v * v; }
    m /= cols; q = q / cols - m * m;
    float inv = rsqrtf(q + LNEPS);
    _Float16* d = D + lane * ldd;
    for (int c = 0; c < cols; ++c) {
      float v = (r[c] - m) * inv * g[c] + bt[c];
      if (dogelu) v = gelu_f(v);
      d[c] = (_Float16)v;
    }
  }
}

/* Et = LN(Et + delta) row-wise, Et stored f16 in LDS                           */
__device__ inline void resid_ln_f16(_Float16* E, int lde, const float* Dl, int ldd,
                                    const float* g, const float* bt, int lane) {
  if (lane < TOK) {
    _Float16* e = E + lane * lde;
    const float* d = Dl + lane * ldd;
    float m = 0.f, q = 0.f;
    for (int c = 0; c < DMODEL; ++c) { float v = (float)e[c] + d[c]; m += v; q += v * v; }
    m /= DMODEL; q = q / DMODEL - m * m;
    float inv = rsqrtf(q + LNEPS);
    for (int c = 0; c < DMODEL; ++c) {
      float v = ((float)e[c] + d[c] - m) * inv * g[c] + bt[c];
      e[c] = (_Float16)v;
    }
  }
}

/* LayerNorm of a 48-vector in LDS via wave32 shuffles                          */
__device__ inline void vec_ln48(const float* v, const float* g, const float* bt,
                                float* dst, int lane) {
  float a0 = v[lane];
  float a1 = (lane < 16) ? v[lane + 32] : 0.0f;
  float s = a0 + a1, q = a0 * a0 + a1 * a1;
  for (int o = 16; o; o >>= 1) { s += __shfl_xor(s, o); q += __shfl_xor(q, o); }
  float m = s * (1.0f / 48.0f);
  float inv = rsqrtf(q * (1.0f / 48.0f) - m * m + LNEPS);
  float r0 = (a0 - m) * inv * g[lane] + bt[lane];
  float r1 = (lane < 16) ? (a1 - m) * inv * g[lane + 32] + bt[lane + 32] : 0.0f;
  dst[lane] = r0;
  if (lane < 16) dst[lane + 32] = r1;
}

/* y[j] = b[j] + sum_k x[k] * Wf[j*K + k]   (single-row matmul, VALU)           */
__device__ inline void vec_mm(const float* x, const float* Wf, const float* b,
                              int N, int K, float* y, int lane) {
  for (int j = lane; j < N; j += 32) {
    float a = b[j];
    const float* wr = Wf + (size_t)j * K;
    for (int k = 0; k < K; ++k) a += x[k] * wr[k];
    y[j] = a;
  }
}

/* self-attention over 22 tokens; QKV in bufC (ld 144: Q|K|V); out f16 (ld 64)  */
__device__ inline void self_attn(const float* QKV, _Float16* Oa, int lane) {
  if (lane >= TOK) return;
  const float scale = 0.2041241452319315f; /* 1/sqrt(24) */
  const float* qr = QKV + lane * 144;
  for (int h = 0; h < NHEAD; ++h) {
    float s[TOK];
    float mx = -1e30f;
#pragma unroll
    for (int t = 0; t < TOK; ++t) {
      const float* kr = QKV + t * 144 + DMODEL + h * HDIM;
      const float* qh = qr + h * HDIM;
      float a = 0.f;
      for (int d = 0; d < HDIM; ++d) a += qh[d] * kr[d];
      a *= scale; s[t] = a; mx = fmaxf(mx, a);
    }
    float sum = 0.f;
#pragma unroll
    for (int t = 0; t < TOK; ++t) { s[t] = expf(s[t] - mx); sum += s[t]; }
    float inv = 1.0f / sum;
    for (int d = 0; d < HDIM; ++d) {
      float acc = 0.f;
#pragma unroll
      for (int t = 0; t < TOK; ++t) acc += s[t] * QKV[t * 144 + 2 * DMODEL + h * HDIM + d];
      Oa[lane * 64 + h * HDIM + d] = (_Float16)(acc * inv);
    }
  }
}

/* single-query cross-attention; KV rows: [K(48) | V(48)], stride ldkv          */
template <typename T>
__device__ inline void cross_attn(const float* q, const T* KV, int ldkv,
                                  float* att, float* o, int lane) {
  const float scale = 0.2041241452319315f;
  for (int h = 0; h < NHEAD; ++h) {
    float s = -1e30f;
    if (lane < TOK) {
      const T* kr = KV + lane * ldkv + h * HDIM;
      float a = 0.f;
      for (int d = 0; d < HDIM; ++d) a += q[h * HDIM + d] * (float)kr[d];
      s = a * scale;
    }
    float mx = s;
    for (int off = 16; off; off >>= 1) mx = fmaxf(mx, __shfl_xor(mx, off));
    float e = (lane < TOK) ? expf(s - mx) : 0.0f;
    float sum = e;
    for (int off = 16; off; off >>= 1) sum += __shfl_xor(sum, off);
    att[lane] = e / sum;
    wsync();
    for (int d = lane; d < HDIM; d += 32) {
      float acc = 0.f;
      for (int t = 0; t < TOK; ++t)
        acc += att[t] * (float)KV[t * ldkv + DMODEL + h * HDIM + d];
      o[h * HDIM + d] = acc;
    }
    wsync();
  }
}

/* ============================ per-wave LDS state ============================ */
struct WLds {
  _Float16 aE[TOK * 64];     /* Et, f16 (A operand, ld 64)                     */
  _Float16 aX[TOK * 96];     /* shared: embed/attn-out (ld 64) | FF hidden (96)*/
  _Float16 e0kv[TOK * 96];   /* precomputed K0|V0 for ce0, f16                 */
  float bufC[32 * 144];      /* f32 matmul outputs; rows 22..31 = scratch      */
  float zv[48], ze0[48], zet[48], yv[48];
  float qt[48], att[32], h96[96], t144[144], t48[48];
};

__global__ __launch_bounds__(64)
void trm_kernel(KParams P, const _Float16* __restrict__ wp,
                float* __restrict__ out, int nsamp) {
  __shared__ WLds lds2[2];
  const int wave = threadIdx.x >> 5;
  const int lane = threadIdx.x & 31;
  int s = blockIdx.x * 2 + wave;
  if (s >= nsamp) s = nsamp - 1;
  WLds& w = lds2[wave];
  const float* const* PR = P.p;
  const float* xr = PR[IDX_X] + (size_t)s * XDIM;

  /* ---- E0 = gelu(LN(magpie @ tp_w.T + tp_b)) ---- */
  if (lane < TOK)
    for (int d = 0; d < SDIM; ++d) w.aX[lane * 64 + d] = (_Float16)xr[lane * SDIM + d];
  wsync();
  mm_wmma(w.aX, 64, SDIM, wp + OFF_TP, 32, DMODEL, PR[IDX_TP_B], w.bufC, 48, lane);
  wsync();
  rows_ln_f16(w.bufC, 48, DMODEL, PR[IDX_TP_G], PR[IDX_TP_BT], true, w.aE, 64, lane);
  wsync();

  /* ---- K0|V0 = E0 @ [wk;wv]_ce0.T + b  (fixed across steps) ---- */
  mm_wmma(w.aE, 64, DMODEL, wp + OFF_CE0KV, 64, 2 * DMODEL,
          PR[IDX_CE0_IN_B] + DMODEL, w.bufC, 96, lane);
  wsync();
  if (lane < TOK)
    for (int c = 0; c < 96; ++c) w.e0kv[lane * 96 + c] = (_Float16)w.bufC[lane * 96 + c];

  /* ---- z0 = gelu(LN(m2v @ zi_w.T + b)) ; y = 0 ---- */
  vec_mm(xr + NPROPS * SDIM, PR[IDX_ZI_W], PR[IDX_ZI_B], 48, M2VD, w.zv, lane);
  wsync();
  vec_ln48(w.zv, PR[IDX_ZI_G], PR[IDX_ZI_BT], w.zv, lane);
  wsync();
  for (int j = lane; j < 48; j += 32) { w.zv[j] = gelu_f(w.zv[j]); w.yv[j] = 0.0f; }
  wsync();

  /* ================================ 16 steps ================================ */
  for (int step = 0; step < NSTEPS; ++step) {
    /* --- Et self-attention block --- */
    mm_wmma(w.aE, 64, DMODEL, wp + OFF_SAIN, 64, 3 * DMODEL,
            PR[IDX_SA_IN_B], w.bufC, 144, lane);
    wsync();
    self_attn(w.bufC, w.aX, lane);
    wsync();
    mm_wmma(w.aX, 64, DMODEL, wp + OFF_SAOUT, 64, DMODEL,
            PR[IDX_SA_OUT_B], w.bufC, 48, lane);
    wsync();
    resid_ln_f16(w.aE, 64, w.bufC, 48, PR[IDX_SAN_G], PR[IDX_SAN_BT], lane);
    wsync();

    /* --- Et feed-forward block --- */
    mm_wmma(w.aE, 64, DMODEL, wp + OFF_FF1, 64, FFD,
            PR[IDX_SAFF1_B], w.bufC, 96, lane);
    wsync();
    if (lane < TOK)
      for (int c = 0; c < FFD; ++c) w.aX[lane * 96 + c] = (_Float16)gelu_f(w.bufC[lane * 96 + c]);
    wsync();
    mm_wmma(w.aX, 96, FFD, wp + OFF_FF2, 96, DMODEL,
            PR[IDX_SAFF2_B], w.bufC, 48, lane);
    wsync();
    resid_ln_f16(w.aE, 64, w.bufC, 48, PR[IDX_SAFFN_G], PR[IDX_SAFFN_BT], lane);
    wsync();

    /* --- Kt|Vt = Et @ [wk;wv]_cet.T + b --- */
    mm_wmma(w.aE, 64, DMODEL, wp + OFF_CETKV, 64, 2 * DMODEL,
            PR[IDX_CET_IN_B] + DMODEL, w.bufC, 96, lane);
    wsync();

    /* --- z_e0 = LN(z + CrossAttn(z, E0)) --- */
    vec_mm(w.zv, PR[IDX_CE0_IN_W], PR[IDX_CE0_IN_B], 48, 48, w.qt, lane);
    wsync();
    cross_attn<_Float16>(w.qt, w.e0kv, 96, w.att, w.t48, lane);
    wsync();
    vec_mm(w.t48, PR[IDX_CE0_OUT_W], PR[IDX_CE0_OUT_B], 48, 48, w.t144, lane);
    wsync();
    for (int j = lane; j < 48; j += 32) w.t144[j] += w.zv[j];
    wsync();
    vec_ln48(w.t144, PR[IDX_CE0N_G], PR[IDX_CE0N_BT], w.ze0, lane);
    wsync();

    /* --- z_et = LN(z + CrossAttn(z, Et)) --- */
    vec_mm(w.zv, PR[IDX_CET_IN_W], PR[IDX_CET_IN_B], 48, 48, w.qt, lane);
    wsync();
    cross_attn<float>(w.qt, w.bufC, 96, w.att, w.t48, lane);
    wsync();
    vec_mm(w.t48, PR[IDX_CET_OUT_W], PR[IDX_CET_OUT_B], 48, 48, w.t144, lane);
    wsync();
    for (int j = lane; j < 48; j += 32) w.t144[j] += w.zv[j];
    wsync();
    vec_ln48(w.t144, PR[IDX_CETN_G], PR[IDX_CETN_BT], w.zet, lane);
    wsync();

    /* --- z += LN(FF([z, z_e0, z_et])) --- */
    for (int j = lane; j < 48; j += 32) {
      w.t144[j] = w.zv[j]; w.t144[48 + j] = w.ze0[j]; w.t144[96 + j] = w.zet[j];
    }
    wsync();
    vec_mm(w.t144, PR[IDX_ZFF1_W], PR[IDX_ZFF1_B], 96, 144, w.h96, lane);
    wsync();
    for (int j = lane; j < 96; j += 32) w.h96[j] = gelu_f(w.h96[j]);
    wsync();
    vec_mm(w.h96, PR[IDX_ZFF2_W], PR[IDX_ZFF2_B], 48, 96, w.t48, lane);
    wsync();
    vec_ln48(w.t48, PR[IDX_ZFFN_G], PR[IDX_ZFFN_BT], w.t48, lane);
    wsync();
    for (int j = lane; j < 48; j += 32) w.zv[j] += w.t48[j];
    wsync();

    /* --- y += LN(FF([y, z])) --- */
    for (int j = lane; j < 48; j += 32) { w.t144[j] = w.yv[j]; w.t144[48 + j] = w.zv[j]; }
    wsync();
    vec_mm(w.t144, PR[IDX_YFF1_W], PR[IDX_YFF1_B], 96, 96, w.h96, lane);
    wsync();
    for (int j = lane; j < 96; j += 32) w.h96[j] = gelu_f(w.h96[j]);
    wsync();
    vec_mm(w.h96, PR[IDX_YFF2_W], PR[IDX_YFF2_B], 48, 96, w.t48, lane);
    wsync();
    vec_ln48(w.t48, PR[IDX_YFFN_G], PR[IDX_YFFN_BT], w.t48, lane);
    wsync();
    for (int j = lane; j < 48; j += 32) w.yv[j] += w.t48[j];
    wsync();
  }

  /* ---- head: out[s] = y . head_w + head_b ---- */
  const float* hw = PR[IDX_HEAD_W];
  float a = w.yv[lane] * hw[lane] + ((lane < 16) ? w.yv[lane + 32] * hw[lane + 32] : 0.0f);
  for (int o = 16; o; o >>= 1) a += __shfl_xor(a, o);
  if (lane == 0) out[s] = a + PR[IDX_HEAD_B][0];
}

/* ================================= launch ================================= */
extern "C" void kernel_launch(void* const* d_in, const int* in_sizes, int n_in,
                              void* d_out, int out_size, void* d_ws, size_t ws_size,
                              hipStream_t stream) {
  (void)out_size; (void)ws_size;
  _Float16* wp = (_Float16*)d_ws;

  struct PackJob { int idx, rowoff, K, N, Kpad, dst; };
  const PackJob jobs[7] = {
    { IDX_TP_W,     0,  SDIM,   DMODEL,     32, OFF_TP    },
    { IDX_SA_IN_W,  0,  DMODEL, 3 * DMODEL, 64, OFF_SAIN  },
    { IDX_SA_OUT_W, 0,  DMODEL, DMODEL,     64, OFF_SAOUT },
    { IDX_SAFF1_W,  0,  DMODEL, FFD,        64, OFF_FF1   },
    { IDX_SAFF2_W,  0,  FFD,    DMODEL,     96, OFF_FF2   },
    { IDX_CE0_IN_W, DMODEL, DMODEL, 2 * DMODEL, 64, OFF_CE0KV },
    { IDX_CET_IN_W, DMODEL, DMODEL, 2 * DMODEL, 64, OFF_CETKV },
  };
  for (int j = 0; j < 7; ++j) {
    int tot = jobs[j].N * jobs[j].Kpad;
    pack_w<<<(tot + 255) / 256, 256, 0, stream>>>(
        (const float*)d_in[jobs[j].idx], jobs[j].rowoff, jobs[j].K, jobs[j].N,
        jobs[j].Kpad, wp + jobs[j].dst);
  }

  KParams P;
  for (int i = 0; i < N_IDX && i < n_in; ++i) P.p[i] = (const float*)d_in[i];

  int n = in_sizes[0] / XDIM;   /* batch size (4096) */
  trm_kernel<<<dim3((n + 1) / 2), dim3(64), 0, stream>>>(P, wp, (float*)d_out, n);
}
